// CrossNat2D_84224308675129
// MI455X (gfx1250) — compile-verified, matrix operand order checked by reading, not run on hardware
//
#include <hip/hip_runtime.h>

typedef __attribute__((ext_vector_type(16))) _Float16 v16h;
typedef __attribute__((ext_vector_type(8)))  float    v8f;

#define BDIM 8
#define HDIM 128
#define WDIM 128
#define CDIM 256
#define MROWS (BDIM * HDIM * WDIM)   // 131072
#define NHEADS 8
#define HD 32
#define KW 7

#define FRAG_STRIDE 20   // 16 halves used + 4 pad (40 B per lane-slot, 8-B aligned)

// fragment slot offset: blk = 16-row (A) / 16-col (B) block, ks = 32-k step, lane
__device__ __forceinline__ int fragOff(int blk, int ks, int lane) {
    return ((blk * 4 + ks) * 32 + lane) * FRAG_STRIDE;
}

// load a fragment (16 contiguous halves) as wide LDS loads
__device__ __forceinline__ v16h ldFrag(const _Float16* p) {
    union { v16h v; uint2 u[4]; } r;
    const uint2* q = (const uint2*)p;
    r.u[0] = q[0]; r.u[1] = q[1]; r.u[2] = q[2]; r.u[3] = q[3];
    return r.v;
}

// f32/f16 -> 4x f16 loaders
__device__ __forceinline__ void load4h(const float* p, _Float16* h) {
    float4 f = *(const float4*)p;
    h[0] = (_Float16)f.x; h[1] = (_Float16)f.y;
    h[2] = (_Float16)f.z; h[3] = (_Float16)f.w;
}
__device__ __forceinline__ void load4h(const _Float16* p, _Float16* h) {
    uint2 u = *(const uint2*)p;
    const _Float16* s = (const _Float16*)&u;
    h[0] = s[0]; h[1] = s[1]; h[2] = s[2]; h[3] = s[3];
}

// ISA A/B 16-bit layout: lane group hi holds K {hi*8..hi*8+7} U {16+hi*8..23+hi*8};
// K pair (k, k+1) lives at fragment halves [2v, 2v+1].
__device__ __forceinline__ void kinToSlot(int Kin, int& hi, int& j) {
    hi = (Kin >> 3) & 1;
    int v = ((Kin & 7) >> 1) + ((Kin & 16) ? 4 : 0);
    j = 2 * v;
}

// ---------------------------------------------------------------------------
// WMMA GEMM: out[M,N] = A[M,256] @ W[256,N] + bias ; cols < scale_ncols get *scale
// Block = 128 threads (4 wave32), tile 64x64, K chunked by 128.
// LDS holds fragment-ready operands -> wide ds loads feed v_wmma.
// ---------------------------------------------------------------------------
template<typename AT, typename OT>
__global__ __launch_bounds__(128)
void wmma_gemm_kernel(const AT* __restrict__ A, const float* __restrict__ Wt,
                      const float* __restrict__ bias, OT* __restrict__ out,
                      int N, float scale, int scale_ncols)
{
    __shared__ _Float16 sAf[4 * 4 * 32 * FRAG_STRIDE];  // 20 KB
    __shared__ _Float16 sBf[4 * 4 * 32 * FRAG_STRIDE];  // 20 KB

    const int tid  = threadIdx.x;
    const int wv   = tid >> 5;
    const int lane = tid & 31;
    const int hi   = lane >> 4;
    const int lc   = lane & 15;
    const long m0  = (long)blockIdx.x * 64;
    const int  n0  = blockIdx.y * 64;

    v8f acc[4] = {};

    for (int kk0 = 0; kk0 < CDIM; kk0 += 128) {
        if (kk0 + 128 < CDIM) {   // prefetch next A chunk (global_prefetch_b8)
            __builtin_prefetch(A + (m0 + (tid >> 1)) * CDIM + kk0 + 128 + (tid & 1) * 64, 0, 0);
        }
        // ---- stage A tile: 64 rows x 128 k, swizzled to fragment layout ----
        for (int e4 = tid; e4 < 64 * 128 / 4; e4 += 128) {
            const int r  = e4 >> 5;
            const int c4 = (e4 & 31) << 2;            // 4 consecutive K
            _Float16 h[4];
            load4h(A + (m0 + r) * CDIM + kk0 + c4, h);
            const int ks = c4 >> 5;
            int hi2, j; kinToSlot(c4 & 31, hi2, j);
            union { uint2 u; _Float16 hh[4]; } pk;
            pk.hh[0] = h[0]; pk.hh[1] = h[1]; pk.hh[2] = h[2]; pk.hh[3] = h[3];
            *(uint2*)&sAf[fragOff(r >> 4, ks, hi2 * 16 + (r & 15)) + j] = pk.u;
        }
        // ---- stage W tile: 128 k x 64 cols; 4 K-strided loads per column so
        //      the 4 halves are one contiguous 8-B LDS store ----
        for (int g = tid; g < 128 * 64 / 4; g += 128) {
            const int col = g & 63;
            const int k0  = (g >> 6) << 2;
            const float* wp = Wt + (long)(kk0 + k0) * N + n0 + col;
            union { uint2 u; _Float16 hh[4]; } pk;
            pk.hh[0] = (_Float16)wp[0];
            pk.hh[1] = (_Float16)wp[(long)N];
            pk.hh[2] = (_Float16)wp[(long)2 * N];
            pk.hh[3] = (_Float16)wp[(long)3 * N];
            const int ks = k0 >> 5;
            int hi2, j; kinToSlot(k0 & 31, hi2, j);
            *(uint2*)&sBf[fragOff(col >> 4, ks, hi2 * 16 + (col & 15)) + j] = pk.u;
        }
        __syncthreads();

        // ---- 4 x (16x16x32) WMMA steps over this 128-k chunk ----
        #pragma unroll
        for (int ks = 0; ks < 4; ++ks) {
            const v16h afrag = ldFrag(&sAf[fragOff(wv, ks, lane)]);
            #pragma unroll
            for (int nb = 0; nb < 4; ++nb) {
                const v16h bfrag = ldFrag(&sBf[fragOff(nb, ks, lane)]);
                acc[nb] = __builtin_amdgcn_wmma_f32_16x16x32_f16(
                    false, afrag, false, bfrag, (short)0, acc[nb], false, false);
            }
        }
        __syncthreads();
    }

    // ---- epilogue: C/D layout row = vgpr + 8*(lane>=16), col = lane&15 ----
    #pragma unroll
    for (int nb = 0; nb < 4; ++nb) {
        const int col = n0 + nb * 16 + lc;
        const float bv = bias[col];
        const float sc = (col < scale_ncols) ? scale : 1.0f;
        #pragma unroll
        for (int r = 0; r < 8; ++r) {
            const long row = m0 + wv * 16 + hi * 8 + r;
            out[row * N + col] = (OT)((acc[nb][r] + bv) * sc);
        }
    }
}

// ---------------------------------------------------------------------------
// Neighborhood attention (7x7 clipped window), 8x8 spatial tile per block.
// K/V tiles (14x14x256 f16, ~100 KB each) + rpb in ~206 KB dynamic LDS
// (CDNA5: 320 KB per workgroup). 512 threads = 64 pixels x 8 heads.
// Tile fill uses GLOBAL_LOAD_ASYNC_TO_LDS_B128 (ASYNCcnt) - no VGPR staging.
// This kernel has NO static LDS, so the dynamic-shared region starts at LDS
// byte offset 0 and async VDST addresses are plain offsets into it.
// qv layout per pixel: [0..255]=q (scaled), [256..511]=v.
// ---------------------------------------------------------------------------
#define SK_BYTES (14 * 14 * CDIM * 2)   // 100352

__global__ __launch_bounds__(512)
void natten_kernel(const _Float16* __restrict__ qv, const _Float16* __restrict__ kbuf,
                   const float* __restrict__ rpb, _Float16* __restrict__ aout)
{
    extern __shared__ char smem[];
    _Float16* sK   = (_Float16*)smem;                 // LDS offset 0
    _Float16* sV   = sK + 14 * 14 * CDIM;             // LDS offset SK_BYTES
    float*    sRpb = (float*)(sV + 14 * 14 * CDIM);   // LDS offset 2*SK_BYTES

    const int tid = threadIdx.x;
    const int blk = blockIdx.x;
    const int tw  = blk & 15;
    const int th2 = (blk >> 4) & 15;
    const int b   = blk >> 8;
    const int i0  = th2 * 8, j0 = tw * 8;
    const int r0  = min(max(i0 - 3, 0), HDIM - KW);
    const int c0  = min(max(j0 - 3, 0), WDIM - KW);

    for (int idx = tid; idx < NHEADS * 13 * 13; idx += 512) sRpb[idx] = rpb[idx];

    // ---- async gather of 14x14 K/V neighborhood straight into LDS ----
    for (int idx = tid; idx < 14 * 14 * (CDIM / 8); idx += 512) {
        const int pix = idx >> 5;
        const int ch8 = (idx & 31) << 3;              // 8 f16 = 16 B
        const int rr  = min(r0 + pix / 14, HDIM - 1);
        const int cc  = min(c0 + pix % 14, WDIM - 1);
        const long mm = ((long)b * HDIM + rr) * WDIM + cc;

        unsigned ldsK = (unsigned)((pix * CDIM + ch8) * 2);
        unsigned ldsV = ldsK + (unsigned)SK_BYTES;
        unsigned long long gk = (unsigned long long)(kbuf + mm * CDIM + ch8);
        unsigned long long gv = (unsigned long long)(qv + mm * (2 * CDIM) + CDIM + ch8);

        asm volatile("global_load_async_to_lds_b128 %0, %1, off"
                     :: "v"(ldsK), "v"(gk) : "memory");
        asm volatile("global_load_async_to_lds_b128 %0, %1, off"
                     :: "v"(ldsV), "v"(gv) : "memory");
    }
    asm volatile("s_wait_asynccnt 0x0" ::: "memory");
    __syncthreads();

    const int p = tid >> 3;
    const int h = tid & 7;
    const int i = i0 + (p >> 3), j = j0 + (p & 7);
    const long m = ((long)b * HDIM + i) * WDIM + j;
    const int si = min(max(i - 3, 0), HDIM - KW);
    const int sj = min(max(j - 3, 0), WDIM - KW);
    const int bi = si - i + (KW - 1);
    const int bj = sj - j + (KW - 1);

    // q (already scaled by hd^-0.5 in GEMM epilogue)
    float q[HD];
    {
        const uint4* qp4 = (const uint4*)(qv + m * (2 * CDIM) + h * HD);
        #pragma unroll
        for (int t = 0; t < 4; ++t) {
            union { uint4 u; _Float16 hh[8]; } w; w.u = qp4[t];
            #pragma unroll
            for (int e = 0; e < 8; ++e) q[t * 8 + e] = (float)w.hh[e];
        }
    }

    // scores + bias (K vectors read as 4x ds_load_b128)
    float sc[KW * KW];
    #pragma unroll
    for (int oi = 0; oi < KW; ++oi) {
        const int krow = si + oi - r0;
        #pragma unroll
        for (int oj = 0; oj < KW; ++oj) {
            const int kcol = sj + oj - c0;
            const uint4* kp4 = (const uint4*)(sK + (krow * 14 + kcol) * CDIM + h * HD);
            float dot = 0.f;
            #pragma unroll
            for (int t = 0; t < 4; ++t) {
                union { uint4 u; _Float16 hh[8]; } w; w.u = kp4[t];
                #pragma unroll
                for (int e = 0; e < 8; ++e) dot += q[t * 8 + e] * (float)w.hh[e];
            }
            sc[oi * KW + oj] = dot + sRpb[h * 169 + (bi + oi) * 13 + (bj + oj)];
        }
    }

    // softmax over 49
    float mx = sc[0];
    #pragma unroll
    for (int n = 1; n < KW * KW; ++n) mx = fmaxf(mx, sc[n]);
    float sum = 0.f;
    #pragma unroll
    for (int n = 0; n < KW * KW; ++n) { sc[n] = __expf(sc[n] - mx); sum += sc[n]; }
    const float inv = 1.f / sum;

    // weighted sum of V (V vectors read as 4x ds_load_b128)
    float acc[HD] = {};
    #pragma unroll
    for (int oi = 0; oi < KW; ++oi) {
        const int krow = si + oi - r0;
        #pragma unroll
        for (int oj = 0; oj < KW; ++oj) {
            const int kcol = sj + oj - c0;
            const float w0 = sc[oi * KW + oj] * inv;
            const uint4* vp4 = (const uint4*)(sV + (krow * 14 + kcol) * CDIM + h * HD);
            #pragma unroll
            for (int t = 0; t < 4; ++t) {
                union { uint4 u; _Float16 hh[8]; } w; w.u = vp4[t];
                #pragma unroll
                for (int e = 0; e < 8; ++e) acc[t * 8 + e] += w0 * (float)w.hh[e];
            }
        }
    }

    // packed f16 output store (4x global_store_b128)
    _Float16* op = aout + m * CDIM + h * HD;
    #pragma unroll
    for (int t = 0; t < 4; ++t) {
        union { uint4 u; _Float16 hh[8]; } o;
        #pragma unroll
        for (int e = 0; e < 8; ++e) o.hh[e] = (_Float16)acc[t * 8 + e];
        ((uint4*)op)[t] = o.u;
    }
}

// ---------------------------------------------------------------------------
// Host-side pipeline
// ---------------------------------------------------------------------------
extern "C" void kernel_launch(void* const* d_in, const int* in_sizes, int n_in,
                              void* d_out, int out_size, void* d_ws, size_t ws_size,
                              hipStream_t stream) {
    const float* x      = (const float*)d_in[0];
    const float* y      = (const float*)d_in[1];
    const float* qv_w   = (const float*)d_in[2];
    const float* qv_b   = (const float*)d_in[3];
    const float* k_w    = (const float*)d_in[4];
    const float* k_b    = (const float*)d_in[5];
    const float* rpb    = (const float*)d_in[6];
    const float* proj_w = (const float*)d_in[7];
    const float* proj_b = (const float*)d_in[8];
    float* out = (float*)d_out;

    char* ws = (char*)d_ws;
    _Float16* qv_f16 = (_Float16*)ws;                                   // M x 512
    _Float16* k_f16  = (_Float16*)(ws + (size_t)MROWS * 512 * 2);       // M x 256
    _Float16* at_f16 = (_Float16*)(ws + (size_t)MROWS * 512 * 2
                                      + (size_t)MROWS * 256 * 2);       // M x 256

    const float qscale = 0.17677669529663687f;  // 32^-0.5
    dim3 gblk(128);

    // qv = x @ qv_w + qv_b  (q cols [0,256) pre-scaled)
    wmma_gemm_kernel<float, _Float16><<<dim3(MROWS / 64, 512 / 64), gblk, 0, stream>>>(
        x, qv_w, qv_b, qv_f16, 512, qscale, 256);
    // k = y @ k_w + k_b
    wmma_gemm_kernel<float, _Float16><<<dim3(MROWS / 64, 256 / 64), gblk, 0, stream>>>(
        y, k_w, k_b, k_f16, 256, 1.0f, 0);

    // neighborhood attention: 2048 tiles, ~206 KB dynamic LDS each
    size_t shmem = (size_t)(14 * 14 * CDIM) * 2 * sizeof(_Float16)
                 + (size_t)NHEADS * 13 * 13 * sizeof(float);
    natten_kernel<<<dim3(BDIM * 16 * 16), dim3(512), shmem, stream>>>(
        qv_f16, k_f16, rpb, at_f16);

    // out = attn @ proj_w + proj_b  (f32 output)
    wmma_gemm_kernel<_Float16, float><<<dim3(MROWS / 64, 256 / 64), gblk, 0, stream>>>(
        at_f16, proj_w, proj_b, out, 256, 1.0f, 0);
}
